// _MoEBlock_52561809768943
// MI455X (gfx1250) — matrix operand order repaired and checked by reading, not compiled, not run on hardware
//
#include <hip/hip_runtime.h>
#include <hip/hip_bf16.h>

// ---------------------------------------------------------------------------
// MoE block: y = x + W[argmax(gate_w @ x)] @ x
// B=2, S=8192 -> 16384 tokens, H=1024, E=8, fp32 in/out.
// Route in fp32 (exact argmax), apply experts via bf16 WMMA (f32 accumulate).
// Grouped GEMM, double-buffered LDS, reg-staged f32->bf16 RNE conversion
// (v_perm-based pack), grouped fragment loads for partial DScnt waits.
// ---------------------------------------------------------------------------

typedef __attribute__((ext_vector_type(16))) __bf16 v16bf;
typedef __attribute__((ext_vector_type(8)))  float  v8f;

#define HID     1024
#define TOKENS  16384
#define NEXP    8
#define TM      64            // block tile M (tokens)
#define TN      128           // block tile N (output features)
#define TKS     64            // K per staging step (2 WMMA sub-steps of 32)
#define KP      72            // LDS row stride in bf16 elems (36 banks)
#define NSTEP   (HID / TKS)   // 16
#define MAX_MTILES (TOKENS / TM + NEXP)   // 264

// RNE f32->bf16 pair pack: 2x (bfe + add3) + 1x v_perm_b32
__device__ __forceinline__ unsigned int pack_bf16x2(float lo, float hi) {
    unsigned int ulo = __float_as_uint(lo);
    unsigned int uhi = __float_as_uint(hi);
    ulo += 0x7FFFu + ((ulo >> 16) & 1u);                 // RNE round in place
    uhi += 0x7FFFu + ((uhi >> 16) & 1u);
    // bytes: D = { uhi[3], uhi[2], ulo[3], ulo[2] }  (src0=uhi -> bytes 4..7)
    return __builtin_amdgcn_perm(uhi, ulo, 0x07060302u);
}

__device__ __forceinline__ uint4 pack8(const float4 a, const float4 b) {
    uint4 r;
    r.x = pack_bf16x2(a.x, a.y);
    r.y = pack_bf16x2(a.z, a.w);
    r.z = pack_bf16x2(b.x, b.y);
    r.w = pack_bf16x2(b.z, b.w);
    return r;
}

// ---------------------------------------------------------------------------
// 0) zero per-expert histogram (workspace is poisoned / stale between calls)
// ---------------------------------------------------------------------------
__global__ void moe_zero_kernel(int* __restrict__ count) {
    if (threadIdx.x < NEXP) count[threadIdx.x] = 0;
}

// ---------------------------------------------------------------------------
// 1) gating: one wave per token; fp32 dot with 8 gate rows from LDS;
//    shuffle-reduce; argmax (first-max, matching jnp.argmax); histogram.
// ---------------------------------------------------------------------------
__global__ __launch_bounds__(256) void moe_gate_kernel(
    const float* __restrict__ x, const float* __restrict__ gate_w,
    int* __restrict__ expert_id, int* __restrict__ count)
{
    __shared__ float gw[NEXP][HID];                         // 32 KB
    {
        const float4* g4 = (const float4*)gate_w;
        float4* s4 = (float4*)&gw[0][0];
        for (int i = threadIdx.x; i < NEXP * HID / 4; i += 256) s4[i] = g4[i];
    }
    __syncthreads();

    const int wave  = threadIdx.x >> 5;
    const int lane  = threadIdx.x & 31;
    const int token = blockIdx.x * 8 + wave;                // 2048 blocks x 8 tokens

    const float4* xr = (const float4*)(x + (size_t)token * HID);
    float acc[NEXP];
#pragma unroll
    for (int e = 0; e < NEXP; ++e) acc[e] = 0.0f;

#pragma unroll
    for (int it = 0; it < HID / (32 * 4); ++it) {           // 8 iters
        const int c4 = it * 32 + lane;                      // float4 index 0..255
        const float4 xv = xr[c4];
#pragma unroll
        for (int e = 0; e < NEXP; ++e) {
            const float4 gv = *(const float4*)&gw[e][c4 * 4];
            acc[e] += xv.x * gv.x + xv.y * gv.y + xv.z * gv.z + xv.w * gv.w;
        }
    }
#pragma unroll
    for (int off = 16; off > 0; off >>= 1)
#pragma unroll
        for (int e = 0; e < NEXP; ++e)
            acc[e] += __shfl_xor(acc[e], off, 32);

    if (lane == 0) {
        float best = acc[0];
        int   be   = 0;
#pragma unroll
        for (int e = 1; e < NEXP; ++e)
            if (acc[e] > best) { best = acc[e]; be = e; }
        expert_id[token] = be;
        atomicAdd(&count[be], 1);
    }
}

// ---------------------------------------------------------------------------
// 2) segment setup: 64-aligned per-expert offsets, m-tile prefix, cursors,
//    sentinel-fill token list (padding rows compute garbage, skip writeback).
// ---------------------------------------------------------------------------
__global__ void moe_setup_kernel(const int* __restrict__ count,
                                 int* __restrict__ segStart,
                                 int* __restrict__ tileBase,
                                 int* __restrict__ cursor,
                                 int* __restrict__ token_list)
{
    if (threadIdx.x == 0) {
        int off = 0, tiles = 0;
        for (int e = 0; e < NEXP; ++e) {
            segStart[e] = off;
            tileBase[e] = tiles;
            cursor[e]   = 0;
            const int ctiles = (count[e] + TM - 1) / TM;
            tiles += ctiles;
            off   += ctiles * TM;
        }
        tileBase[NEXP] = tiles;
    }
    for (int i = threadIdx.x; i < TOKENS + NEXP * TM; i += 256)
        token_list[i] = -1;
}

// ---------------------------------------------------------------------------
// 3) scatter tokens into expert segments (intra-segment order is irrelevant:
//    each token's result is computed independently and written by token id).
// ---------------------------------------------------------------------------
__global__ __launch_bounds__(256) void moe_scatter_kernel(
    const int* __restrict__ expert_id, const int* __restrict__ segStart,
    int* __restrict__ cursor, int* __restrict__ token_list)
{
    const int t = blockIdx.x * 256 + threadIdx.x;
    const int e = expert_id[t];
    const int pos = atomicAdd(&cursor[e], 1);
    token_list[segStart[e] + pos] = t;
}

// ---------------------------------------------------------------------------
// 4) grouped expert GEMM + residual.
//    Y[t, f] = x[t, f] + sum_h bf16(x[t,h]) * bf16(W[e][f,h])
//    64x128 block tile, 8 waves (wave -> 16-col strip, 4x 16x16 accums),
//    double-buffered LDS, K staged 64 at a time (2 WMMA sub-steps).
//    Pipeline: global loads (next) -> 8x WMMA (cur) -> pack+ds_store (next).
// ---------------------------------------------------------------------------
__global__ __launch_bounds__(256) void moe_gemm_kernel(
    const float* __restrict__ x, const float* __restrict__ expert_w,
    const int* __restrict__ token_list, const int* __restrict__ segStart,
    const int* __restrict__ tileBase, float* __restrict__ out)
{
    __shared__ unsigned short As[2][TM][KP];   // 2 x 64 x 72 bf16 = 18 KB
    __shared__ unsigned short Bs[2][TN][KP];   // 2 x 128 x 72 bf16 = 36 KB
    __shared__ int toks[TM];

    const int mt = blockIdx.x;
    if (mt >= tileBase[NEXP]) return;

    int e = 0;
    while (e < NEXP - 1 && mt >= tileBase[e + 1]) ++e;
    const int rowBase = segStart[e] + (mt - tileBase[e]) * TM;
    const int nBase   = blockIdx.y * TN;
    const float* __restrict__ W = expert_w + (size_t)e * HID * HID;

    const int tid  = threadIdx.x;
    const int wv   = tid >> 5;
    const int lane = tid & 31;
    const int lh   = lane >> 4;      // lane half (0/1)
    const int ln   = lane & 15;

    if (tid < TM) toks[tid] = token_list[rowBase + tid];
    __syncthreads();

    // staging geometry (fixed per thread)
    const int rA  = tid >> 2;              // A row 0..63
    const int csA = (tid & 3) * 16;        // A col start (16 k per thread)
    const int fB  = tid >> 1;              // B row (W feature) 0..127
    const int csB = (tid & 1) * 32;        // B col start (32 k per thread)

    const int tokA = toks[rA] < 0 ? 0 : toks[rA];   // branchless pad handling
    const float* __restrict__ xrow = x + (size_t)tokA * HID + csA;
    const float* __restrict__ wrow = W + (size_t)(nBase + fB) * HID + csB;

    float4 aR[4];
    float4 bR[8];

    auto loadTile = [&](int k0) {
#pragma unroll
        for (int i = 0; i < 4; ++i) aR[i] = *(const float4*)(xrow + k0 + 4 * i);
#pragma unroll
        for (int i = 0; i < 8; ++i) bR[i] = *(const float4*)(wrow + k0 + 4 * i);
    };
    auto storeTile = [&](int buf) {
        *(uint4*)&As[buf][rA][csA]     = pack8(aR[0], aR[1]);
        *(uint4*)&As[buf][rA][csA + 8] = pack8(aR[2], aR[3]);
#pragma unroll
        for (int i = 0; i < 4; ++i)
            *(uint4*)&Bs[buf][fB][csB + 8 * i] = pack8(bR[2 * i], bR[2 * i + 1]);
    };

    v8f acc[4];
#pragma unroll
    for (int mi = 0; mi < 4; ++mi) acc[mi] = (v8f){0, 0, 0, 0, 0, 0, 0, 0};

    // prologue: stage tile 0
    loadTile(0);
    storeTile(0);
    __syncthreads();

    for (int ks = 0; ks < NSTEP; ++ks) {
        const int cur = ks & 1;
        const bool more = (ks + 1 < NSTEP);
        if (more) loadTile((ks + 1) * TKS);                  // loads in flight
        if (ks + 2 < NSTEP) {                                // L2/HBM prefetch
            __builtin_prefetch(xrow + (ks + 2) * TKS, 0, 3);
            __builtin_prefetch(wrow + (ks + 2) * TKS, 0, 3);
        }

        // ---- 2 WMMA sub-steps of K=32 from buf[cur] ----
#pragma unroll
        for (int sub = 0; sub < TKS; sub += 32) {
            // Issue ALL fragment loads first (10x ds_load_b128), then the
            // 4 WMMAs: DScnt completes in order, so the scheduler can use
            // partial waits and overlap LDS latency across the WMMA chain.
            union F { v16bf v; uint4 q[2]; };
            F bf;   // B fragment: 32x16 (KxN); lanes 0-15: K lo, 16-31: K hi
            {
                const uint4* bp =
                    (const uint4*)&Bs[cur][(wv << 4) + ln][sub + (lh << 4)];
                bf.q[0] = bp[0];
                bf.q[1] = bp[1];
            }
            F af[4];  // A fragments: 16x32 (MxK); lanes 0-15: K lo, 16-31: K hi
#pragma unroll
            for (int mi = 0; mi < 4; ++mi) {
                const uint4* ap =
                    (const uint4*)&As[cur][(mi << 4) + ln][sub + (lh << 4)];
                af[mi].q[0] = ap[0];
                af[mi].q[1] = ap[1];
            }
#pragma unroll
            for (int mi = 0; mi < 4; ++mi) {
                acc[mi] = __builtin_amdgcn_wmma_f32_16x16x32_bf16(
                    false, af[mi].v, false, bf.v, (short)0, acc[mi],
                    false, false);
            }
        }

        if (more) storeTile(cur ^ 1);                        // behind the WMMAs
        __syncthreads();
    }

    // ---- writeback: out = x + acc ; C layout: VGPR r -> M = r + 8*lh ----
    const int col = nBase + (wv << 4) + ln;
#pragma unroll
    for (int mi = 0; mi < 4; ++mi) {
#pragma unroll
        for (int r = 0; r < 8; ++r) {
            const int m   = (mi << 4) + (lh << 3) + r;
            const int tok = toks[m];
            if (tok >= 0) {
                const size_t idx = (size_t)tok * HID + col;
                out[idx] = x[idx] + acc[mi][r];
            }
        }
    }
}

// ---------------------------------------------------------------------------
// host launcher
// ---------------------------------------------------------------------------
extern "C" void kernel_launch(void* const* d_in, const int* in_sizes, int n_in,
                              void* d_out, int out_size, void* d_ws, size_t ws_size,
                              hipStream_t stream) {
    const float* x        = (const float*)d_in[0];   // [2, 8192, 1024]
    const float* gate_w   = (const float*)d_in[1];   // [8, 1024]
    const float* expert_w = (const float*)d_in[2];   // [8, 1024, 1024]
    float* out = (float*)d_out;

    int* ws        = (int*)d_ws;
    int* count     = ws;                    // 8
    int* cursor    = ws + 8;                // 8
    int* segStart  = ws + 16;               // 8
    int* tileBase  = ws + 24;               // 9
    int* expert_id = ws + 40;               // 16384
    int* token_list= ws + 40 + TOKENS;      // 16384 + 512

    moe_zero_kernel<<<1, 32, 0, stream>>>(count);
    moe_gate_kernel<<<TOKENS / 8, 256, 0, stream>>>(x, gate_w, expert_id, count);
    moe_setup_kernel<<<1, 256, 0, stream>>>(count, segStart, tileBase, cursor, token_list);
    moe_scatter_kernel<<<TOKENS / 256, 256, 0, stream>>>(expert_id, segStart, cursor, token_list);
    moe_gemm_kernel<<<dim3(MAX_MTILES, HID / TN), 256, 0, stream>>>(
        x, expert_w, token_list, segStart, tileBase, out);
}
